// AttnDecoderRNN_33655363731864
// MI455X (gfx1250) — matrix operand order repaired and checked by reading, not compiled, not run on hardware
//
#include <hip/hip_runtime.h>
#include <hip/hip_bf16.h>

// Problem dims (match reference)
#define HD  512
#define VOC 5000
#define LDV 5120   // VOC padded to a multiple of 64 (full WMMA tiles, no guards)
#define BB  128
#define SS  64
#define TT  52

typedef __attribute__((ext_vector_type(16))) __bf16 v16bf;
typedef __attribute__((ext_vector_type(8)))  __bf16 v8bf;
typedef __attribute__((ext_vector_type(8)))  float  v8f;

// ---------------------------------------------------------------------------
// Setup helpers
// ---------------------------------------------------------------------------
__global__ void f32_to_bf16_k(const float* __restrict__ src,
                              __bf16* __restrict__ dst, int n) {
  int i = blockIdx.x * blockDim.x + threadIdx.x;
  if (i < n) dst[i] = (__bf16)src[i];
}

__global__ void bias_sum_k(const float* __restrict__ a,
                           const float* __restrict__ b,
                           float* __restrict__ o, int n) {
  int i = blockIdx.x * blockDim.x + threadIdx.x;
  if (i < n) o[i] = a[i] + b[i];
}

// bout padded to LDV (pad region zeroed; results there are never read)
__global__ void pad_bias_k(const float* __restrict__ src,
                           float* __restrict__ dst, int n, int npad) {
  int i = blockIdx.x * blockDim.x + threadIdx.x;
  if (i < npad) dst[i] = (i < n) ? src[i] : 0.0f;
}

__global__ void init_state_k(const float* __restrict__ h0,
                             const float* __restrict__ c0,
                             float* __restrict__ h, float* __restrict__ c,
                             __bf16* __restrict__ hb) {
  int i = blockIdx.x * blockDim.x + threadIdx.x;
  if (i < BB * HD) {
    h[i] = h0[i];
    c[i] = c0[i];
    hb[i] = (__bf16)h0[i];
  }
}

// ---------------------------------------------------------------------------
// Register-blocked WMMA GEMM core.
// acc[MT][NT] += A[m0.., K] @ W[n0.., K]^T  over K (bf16 in, f32 acc).
// Fragment layout per ISA 7.12.2:
//   A 16x32 bf16: lane half selects K base {0,8} (v0..3) / {16,24} (v4..7),
//                 two contiguous 8-elem (16B) chunks per lane.
//   B 32x16 bf16: same K layout as A: K base {0,8} / {16,24} by half.
//   C/D 16x16 f32: m = v + 8*half, n = lane&15.
// ---------------------------------------------------------------------------
template <int MT, int NT>
__device__ inline void gemm_acc(v8f acc[MT][NT],
                                const __bf16* __restrict__ A,
                                const __bf16* __restrict__ W,
                                int K, int m0, int n0, int half, int l16) {
  const __bf16* Ap[MT];
  const __bf16* Wp[NT];
#pragma unroll
  for (int mt = 0; mt < MT; ++mt)
    Ap[mt] = A + (size_t)(m0 + mt * 16 + l16) * K;
#pragma unroll
  for (int nt = 0; nt < NT; ++nt)
    Wp[nt] = W + (size_t)(n0 + nt * 16 + l16) * K;

  for (int k0 = 0; k0 < K; k0 += 32) {
    // Prefetch the weight stream 256 elements (8 k-blocks) ahead (global_prefetch_b8)
    __builtin_prefetch(Wp[0] + k0 + 256, 0, 1);

    v16bf afr[MT], bfr[NT];
#pragma unroll
    for (int mt = 0; mt < MT; ++mt) {
      v8bf a0 = *(const v8bf*)(Ap[mt] + k0 + half * 8);
      v8bf a1 = *(const v8bf*)(Ap[mt] + k0 + 16 + half * 8);
#pragma unroll
      for (int i = 0; i < 8; ++i) { afr[mt][i] = a0[i]; afr[mt][i + 8] = a1[i]; }
    }
#pragma unroll
    for (int nt = 0; nt < NT; ++nt) {
      v8bf b0 = *(const v8bf*)(Wp[nt] + k0 + half * 8);
      v8bf b1 = *(const v8bf*)(Wp[nt] + k0 + 16 + half * 8);
#pragma unroll
      for (int i = 0; i < 8; ++i) { bfr[nt][i] = b0[i]; bfr[nt][i + 8] = b1[i]; }
    }
#pragma unroll
    for (int mt = 0; mt < MT; ++mt)
#pragma unroll
      for (int nt = 0; nt < NT; ++nt)
        acc[mt][nt] = __builtin_amdgcn_wmma_f32_16x16x32_bf16(
            false, afr[mt], false, bfr[nt], (short)0, acc[mt][nt],
            false, false);
  }
}

// C[M,N] = A1@W1^T (+ A2@W2^T) + bias.  M % (16*MT) == 0, N % (16*NT) == 0,
// K % 32 == 0, row strides == K.  One wave per (16*MT)x(16*NT) macro-tile.
template <int MT, int NT>
__global__ __launch_bounds__(256)
void gemm_wmma_k(const __bf16* __restrict__ A1, const __bf16* __restrict__ W1, int K1,
                 const __bf16* __restrict__ A2, const __bf16* __restrict__ W2, int K2,
                 const float* __restrict__ bias, float* __restrict__ C,
                 int M, int N) {
  const int ntn   = N / (16 * NT);
  const int wave  = (blockIdx.x * blockDim.x + threadIdx.x) >> 5;
  const int total = (M / (16 * MT)) * ntn;
  if (wave >= total) return;   // whole-wave uniform; EXEC all-ones below

  const int m0   = (wave / ntn) * 16 * MT;
  const int n0   = (wave % ntn) * 16 * NT;
  const int lane = threadIdx.x & 31;
  const int half = lane >> 4;
  const int l16  = lane & 15;

  v8f acc[MT][NT];
#pragma unroll
  for (int mt = 0; mt < MT; ++mt)
#pragma unroll
    for (int nt = 0; nt < NT; ++nt) acc[mt][nt] = (v8f){};

  gemm_acc<MT, NT>(acc, A1, W1, K1, m0, n0, half, l16);
  if (K2 > 0) gemm_acc<MT, NT>(acc, A2, W2, K2, m0, n0, half, l16);

#pragma unroll
  for (int nt = 0; nt < NT; ++nt) {
    const int n = n0 + nt * 16 + l16;
    const float bb = bias[n];
#pragma unroll
    for (int mt = 0; mt < MT; ++mt)
#pragma unroll
      for (int v = 0; v < 8; ++v) {
        const int m = m0 + mt * 16 + v + half * 8;
        C[(size_t)m * N + n] = acc[mt][nt][v] + bb;
      }
  }
}

// ---------------------------------------------------------------------------
// Fused attention step: scores -> softmax -> context + embedding gather.
// One 256-thread block per batch row b.
// ---------------------------------------------------------------------------
__global__ __launch_bounds__(256)
void attn_fused_k(const float* __restrict__ q, const float* __restrict__ keysp,
                  const float* __restrict__ Va, const float* __restrict__ bv,
                  const float* __restrict__ enc, const int* __restrict__ target,
                  const __bf16* __restrict__ emb_b, __bf16* __restrict__ xctx,
                  float* __restrict__ out_attn, int t) {
  __shared__ float sc[SS];
  const int b = blockIdx.x;
  const int tid = threadIdx.x;
  const int wave = tid >> 5, lane = tid & 31;
  const float* qp = q + (size_t)b * HD;

  // scores[b,s] = Va . tanh(q + keys_proj) + bv   (one wave per 8 s-values)
  for (int s = wave; s < SS; s += 8) {
    const float* kp = keysp + ((size_t)b * SS + s) * HD;
    float v = 0.f;
    for (int h = lane; h < HD; h += 32)
      v += tanhf(qp[h] + kp[h]) * Va[h];
    for (int off = 16; off; off >>= 1) v += __shfl_xor(v, off, 32);
    if (lane == 0) sc[s] = v + bv[0];
  }
  __syncthreads();

  // softmax over S=64 (wave 0), write attn output + weights back to LDS
  if (wave == 0) {
    float s0 = sc[lane], s1 = sc[lane + 32];
    float m = fmaxf(s0, s1);
    for (int off = 16; off; off >>= 1) m = fmaxf(m, __shfl_xor(m, off, 32));
    float e0 = expf(s0 - m), e1 = expf(s1 - m);
    float sum = e0 + e1;
    for (int off = 16; off; off >>= 1) sum += __shfl_xor(sum, off, 32);
    float inv = 1.f / sum;
    float w0 = e0 * inv, w1 = e1 * inv;
    sc[lane] = w0; sc[lane + 32] = w1;
    size_t base = ((size_t)b * TT + t) * SS;
    out_attn[base + lane] = w0;
    out_attn[base + lane + 32] = w1;
  }
  __syncthreads();

  // ctx[b,h] = sum_s w[s]*enc[b,s,h]  -> xctx[:, H:2H];  x = emb[tok] -> xctx[:, :H]
  const int tok = (t == 0) ? 0 : target[b * TT + (t - 1)];
  for (int h = tid; h < HD; h += 256) {
    const float* ep = enc + ((size_t)b * SS) * HD + h;
    float acc = 0.f;
#pragma unroll 8
    for (int s = 0; s < SS; ++s) acc += sc[s] * ep[(size_t)s * HD];
    xctx[(size_t)b * (2 * HD) + HD + h] = (__bf16)acc;
    xctx[(size_t)b * (2 * HD) + h] = emb_b[(size_t)tok * HD + h];
  }
}

// ---------------------------------------------------------------------------
// LSTM cell update (torch gate order i,f,g,o); refresh h (f32 + bf16) and c.
// ---------------------------------------------------------------------------
__global__ void cell_k(const float* __restrict__ gates,
                       float* __restrict__ h, float* __restrict__ c,
                       __bf16* __restrict__ hb) {
  int i = blockIdx.x * blockDim.x + threadIdx.x;
  if (i >= BB * HD) return;
  int b = i / HD, hh = i % HD;
  const float* g = gates + (size_t)b * 4 * HD;
  float ig = g[hh], fg = g[HD + hh], gg = g[2 * HD + hh], og = g[3 * HD + hh];
  float si = 1.f / (1.f + expf(-ig));
  float sf = 1.f / (1.f + expf(-fg));
  float so = 1.f / (1.f + expf(-og));
  float cn = sf * c[i] + si * tanhf(gg);
  float hn = so * tanhf(cn);
  c[i] = cn;
  h[i] = hn;
  hb[i] = (__bf16)hn;
}

// ---------------------------------------------------------------------------
// log_softmax over V=5000 (row stride LDV); one 256-thread block per row.
// ---------------------------------------------------------------------------
__global__ __launch_bounds__(256)
void logsoftmax_k(const float* __restrict__ logits,
                  float* __restrict__ out, int t) {
  __shared__ float red[8];
  const int b = blockIdx.x;
  const int tid = threadIdx.x;
  const float* lp = logits + (size_t)b * LDV;

  float m = -3.0e38f;
  for (int v = tid; v < VOC; v += 256) m = fmaxf(m, lp[v]);
  for (int off = 16; off; off >>= 1) m = fmaxf(m, __shfl_xor(m, off, 32));
  if ((tid & 31) == 0) red[tid >> 5] = m;
  __syncthreads();
  if (tid < 8) {
    float x = red[tid];
    for (int off = 4; off; off >>= 1) x = fmaxf(x, __shfl_xor(x, off, 32));
    if (tid == 0) red[0] = x;
  }
  __syncthreads();
  m = red[0];
  __syncthreads();

  float sum = 0.f;
  for (int v = tid; v < VOC; v += 256) sum += expf(lp[v] - m);
  for (int off = 16; off; off >>= 1) sum += __shfl_xor(sum, off, 32);
  if ((tid & 31) == 0) red[tid >> 5] = sum;
  __syncthreads();
  if (tid < 8) {
    float x = red[tid];
    for (int off = 4; off; off >>= 1) x += __shfl_xor(x, off, 32);
    if (tid == 0) red[0] = x;
  }
  __syncthreads();
  const float lse = m + logf(red[0]);

  float* op = out + ((size_t)b * TT + t) * VOC;
  for (int v = tid; v < VOC; v += 256) op[v] = lp[v] - lse;
}

__global__ void final_hc_k(const float* __restrict__ h,
                           const float* __restrict__ c,
                           float* __restrict__ out) {
  int i = blockIdx.x * blockDim.x + threadIdx.x;
  if (i >= BB * HD) return;
  const size_t LOGP = (size_t)BB * TT * VOC;
  out[LOGP + i] = h[i];
  out[LOGP + (size_t)BB * HD + i] = c[i];
}

// ---------------------------------------------------------------------------
// Host orchestration
// ---------------------------------------------------------------------------
extern "C" void kernel_launch(void* const* d_in, const int* in_sizes, int n_in,
                              void* d_out, int out_size, void* d_ws, size_t ws_size,
                              hipStream_t stream) {
  const float* enc  = (const float*)d_in[0];
  const float* h0   = (const float*)d_in[1];
  const float* c0   = (const float*)d_in[2];
  const int*   tgt  = (const int*)  d_in[3];
  const float* emb  = (const float*)d_in[4];
  const float* Wa   = (const float*)d_in[5];
  const float* ba   = (const float*)d_in[6];
  const float* Ua   = (const float*)d_in[7];
  const float* bu   = (const float*)d_in[8];
  const float* Va   = (const float*)d_in[9];
  const float* bv   = (const float*)d_in[10];
  const float* Wih  = (const float*)d_in[11];
  const float* Whh  = (const float*)d_in[12];
  const float* bih  = (const float*)d_in[13];
  const float* bhh  = (const float*)d_in[14];
  const float* Wout = (const float*)d_in[15];
  const float* bout = (const float*)d_in[16];
  float* out = (float*)d_out;

  // Workspace bump allocator (256B aligned regions)
  char* p = (char*)d_ws;
  auto alloc = [&](size_t bytes) -> char* {
    char* r = p;
    p += (bytes + 255) & ~(size_t)255;
    return r;
  };
  __bf16* enc_b   = (__bf16*)alloc((size_t)BB * SS * HD * 2);
  __bf16* emb_b   = (__bf16*)alloc((size_t)VOC * HD * 2);
  __bf16* Wa_b    = (__bf16*)alloc((size_t)HD * HD * 2);
  __bf16* Ua_b    = (__bf16*)alloc((size_t)HD * HD * 2);
  __bf16* Wih_b   = (__bf16*)alloc((size_t)4 * HD * 2 * HD * 2);
  __bf16* Whh_b   = (__bf16*)alloc((size_t)4 * HD * HD * 2);
  __bf16* Wout_b  = (__bf16*)alloc((size_t)LDV * HD * 2);   // rows 5000..5119 unused
  float*  bout_p  = (float*) alloc((size_t)LDV * 4);
  float*  keysp   = (float*) alloc((size_t)BB * SS * HD * 4);
  float*  qbuf    = (float*) alloc((size_t)BB * HD * 4);
  __bf16* xctx    = (__bf16*)alloc((size_t)BB * 2 * HD * 2);
  float*  gates   = (float*) alloc((size_t)BB * 4 * HD * 4);
  float*  hbuf    = (float*) alloc((size_t)BB * HD * 4);
  float*  cbuf    = (float*) alloc((size_t)BB * HD * 4);
  __bf16* h_b     = (__bf16*)alloc((size_t)BB * HD * 2);
  float*  bsum    = (float*) alloc((size_t)4 * HD * 4);
  float*  logits  = (float*) alloc((size_t)BB * LDV * 4);

  auto cvt = [&](const float* s, __bf16* d, int n) {
    f32_to_bf16_k<<<(n + 255) / 256, 256, 0, stream>>>(s, d, n);
  };

  // ---- One-time setup ----
  cvt(enc,  enc_b,  BB * SS * HD);
  cvt(emb,  emb_b,  VOC * HD);
  cvt(Wa,   Wa_b,   HD * HD);
  cvt(Ua,   Ua_b,   HD * HD);
  cvt(Wih,  Wih_b,  4 * HD * 2 * HD);
  cvt(Whh,  Whh_b,  4 * HD * HD);
  cvt(Wout, Wout_b, VOC * HD);
  pad_bias_k<<<(LDV + 255) / 256, 256, 0, stream>>>(bout, bout_p, VOC, LDV);
  bias_sum_k<<<(4 * HD + 255) / 256, 256, 0, stream>>>(bih, bhh, bsum, 4 * HD);
  init_state_k<<<(BB * HD + 255) / 256, 256, 0, stream>>>(h0, c0, hbuf, cbuf, h_b);

  // keys_proj = enc @ Ua^T + bu : M=B*S=8192, N=512, K=512 (64x64 macro-tiles)
  {
    int tiles = (BB * SS / 64) * (HD / 64);
    gemm_wmma_k<4, 4><<<(tiles + 7) / 8, 256, 0, stream>>>(
        enc_b, Ua_b, HD, nullptr, nullptr, 0, bu, keysp, BB * SS, HD);
  }

  const int g_bh = (BB * HD + 255) / 256;
  float* out_attn = out + (size_t)BB * TT * VOC + 2 * (size_t)BB * HD;

  // ---- Sequential decode: 52 steps, 6 launches each ----
  for (int t = 0; t < TT; ++t) {
    // q = h @ Wa^T + ba : 32x32 macro-tiles -> 64 waves
    {
      int tiles = (BB / 32) * (HD / 32);
      gemm_wmma_k<2, 2><<<(tiles + 7) / 8, 256, 0, stream>>>(
          h_b, Wa_b, HD, nullptr, nullptr, 0, ba, qbuf, BB, HD);
    }

    attn_fused_k<<<BB, 256, 0, stream>>>(qbuf, keysp, Va, bv, enc, tgt,
                                         emb_b, xctx, out_attn, t);

    // gates = [x,ctx] @ W_ih^T + h @ W_hh^T + (b_ih+b_hh) : dual-K fused
    {
      int tiles = (BB / 32) * (4 * HD / 64);
      gemm_wmma_k<2, 4><<<(tiles + 7) / 8, 256, 0, stream>>>(
          xctx, Wih_b, 2 * HD, h_b, Whh_b, HD, bsum, gates, BB, 4 * HD);
    }

    cell_k<<<g_bh, 256, 0, stream>>>(gates, hbuf, cbuf, h_b);

    // logits = h @ Wout^T + bout (N padded to 5120)
    {
      int tiles = (BB / 32) * (LDV / 64);
      gemm_wmma_k<2, 4><<<(tiles + 7) / 8, 256, 0, stream>>>(
          h_b, Wout_b, HD, nullptr, nullptr, 0, bout_p, logits, BB, LDV);
    }

    logsoftmax_k<<<BB, 256, 0, stream>>>(logits, out, t);
  }

  final_hc_k<<<g_bh, 256, 0, stream>>>(hbuf, cbuf, out);
}
